// MMDiTBlock_47863115546865
// MI455X (gfx1250) — matrix operand order repaired and checked by reading, not compiled, not run on hardware
//
#include <hip/hip_runtime.h>

typedef __attribute__((ext_vector_type(16))) __bf16 v16bf;
typedef __attribute__((ext_vector_type(8)))  float  v8f;

union Frag { v16bf v; uint4 q[2]; };

static __device__ inline v8f v8f_zero() {
    v8f v;
    #pragma unroll
    for (int i = 0; i < 8; ++i) v[i] = 0.0f;
    return v;
}

// fp32 -> bf16 (round to nearest even)
static __device__ inline unsigned short f2bf(float f) {
    union { float f; unsigned int u; } v; v.f = f;
    unsigned int r = v.u + 0x7fffu + ((v.u >> 16) & 1u);
    return (unsigned short)(r >> 16);
}
static __device__ inline float bf2f(unsigned short h) {
    union { unsigned int u; float f; } v; v.u = ((unsigned int)h) << 16;
    return v.f;
}

#define C_DIM   1024
#define SIXC    6144
#define NT      1280
#define NP      1024
#define HEADS   16
#define DH      64
#define HID     2730
#define HIDP    2752      // HID padded to 32
#define N2      5460      // 2*HID
#define N2P     5504      // 2*HID padded to 128

// CDNA5 async global->LDS copy (ASYNCcnt-tracked DMA, no VGPR round trip)
#if defined(__has_builtin)
#if __has_builtin(__builtin_amdgcn_global_load_async_to_lds_b128)
#define HAVE_ASYNC_LDS 1
#endif
#endif
#ifndef HAVE_ASYNC_LDS
#define HAVE_ASYNC_LDS 0
#endif

#if HAVE_ASYNC_LDS
typedef int v4i_t __attribute__((__vector_size__(16)));
static __device__ inline void async_cp16(const unsigned short* g, unsigned short* l) {
    __builtin_amdgcn_global_load_async_to_lds_b128(
        (__attribute__((address_space(1))) v4i_t*)g,
        (__attribute__((address_space(3))) v4i_t*)l, 0, 0);
}
#endif

// ---------------------------------------------------------------------------
// Weight convert: w[K][N] fp32 -> wt[Nalloc][Kalloc] bf16 (transposed, padded)
// ---------------------------------------------------------------------------
__global__ void mmd_convw(const float* __restrict__ w, unsigned short* __restrict__ wt,
                          int N, int K, int Nalloc, int Kalloc) {
    size_t idx = (size_t)blockIdx.x * 256 + threadIdx.x;
    size_t total = (size_t)Nalloc * Kalloc;
    if (idx >= total) return;
    int k = (int)(idx % Kalloc);
    int n = (int)(idx / Kalloc);
    unsigned short v = 0;
    if (n < N && k < K) v = f2bf(w[(size_t)k * N + n]);
    wt[idx] = v;
}

// ---------------------------------------------------------------------------
// adaLN: ada = silu(c) @ w_ada + b_ada  (two matrices, M=4, tiny -> VALU)
// grid (48, 4), block 256
// ---------------------------------------------------------------------------
__global__ __launch_bounds__(256) void mmd_ada(
    const float* __restrict__ c,
    const float* __restrict__ w_p, const float* __restrict__ b_p,
    const float* __restrict__ w_d, const float* __restrict__ b_d,
    float* __restrict__ ada_p, float* __restrict__ ada_d) {
    __shared__ float sc[C_DIM];
    int b = blockIdx.y;
    int tid = threadIdx.x;
    for (int i = tid; i < C_DIM; i += 256) {
        float v = c[b * C_DIM + i];
        sc[i] = v / (1.0f + __expf(-v));
    }
    __syncthreads();
    int which = blockIdx.x / 24;
    int j = (blockIdx.x % 24) * 256 + tid;
    const float* w  = which ? w_d : w_p;
    const float* bb = which ? b_d : b_p;
    float acc = 0.0f;
    for (int k = 0; k < C_DIM; ++k) acc += sc[k] * w[(size_t)k * SIXC + j];
    float* o = which ? ada_d : ada_p;
    o[(size_t)b * SIXC + j] = acc + bb[j];
}

// ---------------------------------------------------------------------------
// RMSNorm + modulate -> bf16 row.  grid = B*tok blocks, block 256
// ---------------------------------------------------------------------------
__global__ __launch_bounds__(256) void mmd_normmod(
    const float* __restrict__ x, int tok,
    const float* __restrict__ gamma,
    const float* __restrict__ ada, int shOff, int scOff,
    unsigned short* __restrict__ out, int outTokPerB, int outTokOff) {
    __shared__ float red[256];
    int bi = blockIdx.x;
    int b = bi / tok, n = bi % tok;
    const float* xr = x + (size_t)bi * C_DIM;
    int tid = threadIdx.x;
    float vals[4];
    float ss = 0.0f;
    #pragma unroll
    for (int i = 0; i < 4; ++i) { vals[i] = xr[tid + i * 256]; ss += vals[i] * vals[i]; }
    red[tid] = ss; __syncthreads();
    for (int s = 128; s >= 1; s >>= 1) { if (tid < s) red[tid] += red[tid + s]; __syncthreads(); }
    float r = rsqrtf(red[0] * (1.0f / C_DIM) + 1e-6f);
    unsigned short* orow = out + ((size_t)(b * outTokPerB + outTokOff + n)) * C_DIM;
    const float* adab = ada + (size_t)b * SIXC;
    #pragma unroll
    for (int i = 0; i < 4; ++i) {
        int cc = tid + i * 256;
        float v = vals[i] * r * gamma[cc];
        v = v * (1.0f + adab[scOff + cc]) + adab[shOff + cc];
        orow[cc] = f2bf(v);
    }
}

// ---------------------------------------------------------------------------
// Tiled WMMA GEMM: C = A(bf16,[M][K]) * Bt(bf16,[N][K])^T + bias
// block 256 (8 waves), tile 128x128, K-step 32; wave tile 64x32.
// Double-buffered LDS; async global->LDS DMA when available.
// ---------------------------------------------------------------------------
__global__ __launch_bounds__(256) void mmd_gemm(
    const unsigned short* __restrict__ A,
    const unsigned short* __restrict__ Bt,
    const float* __restrict__ bias,
    float* __restrict__ outF, unsigned short* __restrict__ outB,
    int M, int N, int K, int ldc) {
    __shared__ __align__(16) unsigned short As[2][128 * 32];
    __shared__ __align__(16) unsigned short Bs[2][128 * 32];
    const int tid = threadIdx.x;
    const int lane = tid & 31, w = tid >> 5;
    const int wm = w & 1, wn = w >> 1;
    const int lm = lane & 15, lh = lane >> 4;
    const int rowBase = blockIdx.y * 128;
    const int colBase = blockIdx.x * 128;
    const int ldRow = tid >> 1;
    const int ldK = (tid & 1) * 16;
    v8f acc[4][2];
    #pragma unroll
    for (int i = 0; i < 4; ++i)
        #pragma unroll
        for (int j = 0; j < 2; ++j) acc[i][j] = v8f_zero();
    const unsigned short* gA = A  + (size_t)(rowBase + ldRow) * K + ldK;
    const unsigned short* gB = Bt + (size_t)(colBase + ldRow) * K + ldK;
    const int ldsOff = ldRow * 32 + ldK;

    // prologue: fill buffer 0 with K-step 0
#if HAVE_ASYNC_LDS
    async_cp16(gA,     &As[0][ldsOff]);
    async_cp16(gA + 8, &As[0][ldsOff + 8]);
    async_cp16(gB,     &Bs[0][ldsOff]);
    async_cp16(gB + 8, &Bs[0][ldsOff + 8]);
    asm volatile("s_wait_asynccnt 0x0" ::: "memory");
#else
    {
        uint4 a0 = *(const uint4*)(gA);
        uint4 a1 = *(const uint4*)(gA + 8);
        uint4 b0 = *(const uint4*)(gB);
        uint4 b1 = *(const uint4*)(gB + 8);
        *(uint4*)&As[0][ldsOff]     = a0;
        *(uint4*)&As[0][ldsOff + 8] = a1;
        *(uint4*)&Bs[0][ldsOff]     = b0;
        *(uint4*)&Bs[0][ldsOff + 8] = b1;
    }
#endif
    __syncthreads();

    int cur = 0;
    const int nsteps = K >> 5;
    for (int it = 0; it < nsteps; ++it) {
        const int knext = (it + 1) << 5;
        const bool hasNext = knext < K;
        const int nxt = cur ^ 1;

#if HAVE_ASYNC_LDS
        if (hasNext) {
            async_cp16(gA + knext,     &As[nxt][ldsOff]);
            async_cp16(gA + knext + 8, &As[nxt][ldsOff + 8]);
            async_cp16(gB + knext,     &Bs[nxt][ldsOff]);
            async_cp16(gB + knext + 8, &Bs[nxt][ldsOff + 8]);
        }
#else
        uint4 a0, a1, b0, b1;
        if (hasNext) {
            a0 = *(const uint4*)(gA + knext);
            a1 = *(const uint4*)(gA + knext + 8);
            b0 = *(const uint4*)(gB + knext);
            b1 = *(const uint4*)(gB + knext + 8);
        }
#endif
        // compute on current buffer while next tile is in flight
        Frag af[4], bfr[2];
        #pragma unroll
        for (int mt = 0; mt < 4; ++mt) {
            int m = wm * 64 + mt * 16 + lm;
            af[mt].q[0] = *(const uint4*)&As[cur][m * 32 + lh * 8];
            af[mt].q[1] = *(const uint4*)&As[cur][m * 32 + 16 + lh * 8];
        }
        #pragma unroll
        for (int nt = 0; nt < 2; ++nt) {
            int n = wn * 32 + nt * 16 + lm;
            bfr[nt].q[0] = *(const uint4*)&Bs[cur][n * 32 + lh * 8];
            bfr[nt].q[1] = *(const uint4*)&Bs[cur][n * 32 + 16 + lh * 8];
        }
        #pragma unroll
        for (int mt = 0; mt < 4; ++mt)
            #pragma unroll
            for (int nt = 0; nt < 2; ++nt)
                acc[mt][nt] = __builtin_amdgcn_wmma_f32_16x16x32_bf16(
                    false, af[mt].v, false, bfr[nt].v, (short)0, acc[mt][nt], false, false);

#if HAVE_ASYNC_LDS
        if (hasNext) asm volatile("s_wait_asynccnt 0x0" ::: "memory");
#else
        if (hasNext) {
            *(uint4*)&As[nxt][ldsOff]     = a0;
            *(uint4*)&As[nxt][ldsOff + 8] = a1;
            *(uint4*)&Bs[nxt][ldsOff]     = b0;
            *(uint4*)&Bs[nxt][ldsOff + 8] = b1;
        }
#endif
        __syncthreads();
        cur = nxt;
    }

    #pragma unroll
    for (int mt = 0; mt < 4; ++mt) {
        #pragma unroll
        for (int nt = 0; nt < 2; ++nt) {
            int col = colBase + wn * 32 + nt * 16 + lm;
            if (col >= N) continue;
            float bv = bias ? bias[col] : 0.0f;
            #pragma unroll
            for (int r = 0; r < 8; ++r) {
                int row = rowBase + wm * 64 + mt * 16 + lh * 8 + r;
                float v = acc[mt][nt][r] + bv;
                if (outF) outF[(size_t)row * ldc + col] = v;
                else      outB[(size_t)row * ldc + col] = f2bf(v);
            }
        }
    }
}

// ---------------------------------------------------------------------------
// qkv fp32 [B,NT,3C] -> qk-RMSNorm + RoPE -> q,k bf16 [B,H,N,DH]; v -> vt [B,H,DH,N]
// one wave per (b,h,n)
// ---------------------------------------------------------------------------
__global__ __launch_bounds__(32) void mmd_qkv_prep(
    const float* __restrict__ qkv,
    const float* __restrict__ gqn, const float* __restrict__ gkn,
    unsigned short* __restrict__ qb, unsigned short* __restrict__ kb,
    unsigned short* __restrict__ vt) {
    int gid = blockIdx.x;                 // (b*16+h)*1280 + n
    int n = gid % NT;
    int bh = gid / NT;
    int h = bh % HEADS;
    int t = threadIdx.x;                  // 0..31
    const float* row = qkv + (size_t)(bh / HEADS * NT + n) * (3 * C_DIM);
    float pos = (float)(n < NP ? n : n - NP);
    float ang = pos * __expf(-9.210340371976184f * (float)t * (1.0f / 32.0f));
    float cs = __cosf(ang), sn = __sinf(ang);
    {   // Q
        float x1 = row[h * DH + t], x2 = row[h * DH + t + 32];
        float ss = x1 * x1 + x2 * x2;
        for (int mk = 1; mk < 32; mk <<= 1) ss += __shfl_xor(ss, mk, 32);
        float r = rsqrtf(ss * (1.0f / DH) + 1e-6f);
        float n1 = x1 * r * gqn[t], n2 = x2 * r * gqn[t + 32];
        unsigned short* qd = qb + ((size_t)bh * NT + n) * DH;
        qd[t]      = f2bf(n1 * cs - n2 * sn);
        qd[t + 32] = f2bf(n1 * sn + n2 * cs);
    }
    {   // K
        float x1 = row[C_DIM + h * DH + t], x2 = row[C_DIM + h * DH + t + 32];
        float ss = x1 * x1 + x2 * x2;
        for (int mk = 1; mk < 32; mk <<= 1) ss += __shfl_xor(ss, mk, 32);
        float r = rsqrtf(ss * (1.0f / DH) + 1e-6f);
        float n1 = x1 * r * gkn[t], n2 = x2 * r * gkn[t + 32];
        unsigned short* kd = kb + ((size_t)bh * NT + n) * DH;
        kd[t]      = f2bf(n1 * cs - n2 * sn);
        kd[t + 32] = f2bf(n1 * sn + n2 * cs);
    }
    {   // V transposed
        unsigned short* vd = vt + (size_t)bh * DH * NT;
        vd[(size_t)t * NT + n]        = f2bf(row[2 * C_DIM + h * DH + t]);
        vd[(size_t)(t + 32) * NT + n] = f2bf(row[2 * C_DIM + h * DH + t + 32]);
    }
}

// ---------------------------------------------------------------------------
// Flash attention with WMMA. grid (10, H, B), block 256 (8 waves x 16 q-rows)
// ---------------------------------------------------------------------------
__global__ __launch_bounds__(256) void mmd_attn(
    const unsigned short* __restrict__ qb,
    const unsigned short* __restrict__ kb,
    const unsigned short* __restrict__ vt,
    const unsigned char* __restrict__ blockFlag,
    unsigned short* __restrict__ attnOut) {
    __shared__ __align__(16) unsigned short Pbuf[8][16 * 32];
    const int tid = threadIdx.x, lane = tid & 31, w = tid >> 5;
    const int lm = lane & 15, lh = lane >> 4;
    const int b = blockIdx.z, h = blockIdx.y;
    const int qbase = blockIdx.x * 128 + w * 16;
    const size_t hb = (size_t)(b * HEADS + h);
    const unsigned short* Q  = qb + hb * NT * DH;
    const unsigned short* Kp = kb + hb * NT * DH;
    const unsigned short* Vp = vt + hb * DH * NT;
    const bool blocked = blockFlag[b] != 0;

    Frag aq[2];
    {
        const unsigned short* qp = Q + (size_t)(qbase + lm) * DH;
        #pragma unroll
        for (int f = 0; f < 2; ++f) {
            aq[f].q[0] = *(const uint4*)(qp + f * 32 + lh * 8);
            aq[f].q[1] = *(const uint4*)(qp + f * 32 + 16 + lh * 8);
        }
    }
    float mrow[8], lrow[8];
    v8f accO[4];
    #pragma unroll
    for (int r = 0; r < 8; ++r) { mrow[r] = -1e30f; lrow[r] = 0.0f; }
    #pragma unroll
    for (int dt = 0; dt < 4; ++dt) accO[dt] = v8f_zero();
    bool qpix[8];
    #pragma unroll
    for (int r = 0; r < 8; ++r) qpix[r] = (qbase + lh * 8 + r) < NP;
    unsigned short* Pw = &Pbuf[w][0];

    for (int kt = 0; kt < NT / 32; ++kt) {
        v8f st[2];
        #pragma unroll
        for (int s = 0; s < 2; ++s) {
            int key = kt * 32 + s * 16 + lm;
            const unsigned short* kp = Kp + (size_t)key * DH;
            Frag bk0, bk1;
            bk0.q[0] = *(const uint4*)(kp + lh * 8);
            bk0.q[1] = *(const uint4*)(kp + 16 + lh * 8);
            bk1.q[0] = *(const uint4*)(kp + 32 + lh * 8);
            bk1.q[1] = *(const uint4*)(kp + 32 + 16 + lh * 8);
            v8f sacc = v8f_zero();
            sacc = __builtin_amdgcn_wmma_f32_16x16x32_bf16(false, aq[0].v, false, bk0.v, (short)0, sacc, false, false);
            sacc = __builtin_amdgcn_wmma_f32_16x16x32_bf16(false, aq[1].v, false, bk1.v, (short)0, sacc, false, false);
            bool kd = blocked && (key >= NP);
            #pragma unroll
            for (int r = 0; r < 8; ++r) {
                float v = sacc[r] * 0.125f;           // 1/sqrt(64)
                if (kd && qpix[r]) v = -1e30f;
                sacc[r] = v;
            }
            st[s] = sacc;
        }
        #pragma unroll
        for (int r = 0; r < 8; ++r) {
            float tmax = fmaxf(st[0][r], st[1][r]);
            for (int mk = 1; mk < 16; mk <<= 1) tmax = fmaxf(tmax, __shfl_xor(tmax, mk, 32));
            float nm = fmaxf(mrow[r], tmax);
            float fs = __expf(mrow[r] - nm);
            float p0 = __expf(st[0][r] - nm);
            float p1 = __expf(st[1][r] - nm);
            float rs = p0 + p1;
            for (int mk = 1; mk < 16; mk <<= 1) rs += __shfl_xor(rs, mk, 32);
            lrow[r] = lrow[r] * fs + rs;
            mrow[r] = nm;
            #pragma unroll
            for (int dt = 0; dt < 4; ++dt) accO[dt][r] *= fs;
            int m = lh * 8 + r;
            Pw[m * 32 + lm]      = f2bf(p0);
            Pw[m * 32 + 16 + lm] = f2bf(p1);
        }
        asm volatile("s_wait_dscnt 0x0" ::: "memory");
        Frag pa;
        pa.q[0] = *(const uint4*)(Pw + lm * 32 + lh * 8);
        pa.q[1] = *(const uint4*)(Pw + lm * 32 + 16 + lh * 8);
        #pragma unroll
        for (int dt = 0; dt < 4; ++dt) {
            int dcol = dt * 16 + lm;
            const unsigned short* vp = Vp + (size_t)dcol * NT + kt * 32;
            Frag bv;
            bv.q[0] = *(const uint4*)(vp + lh * 8);
            bv.q[1] = *(const uint4*)(vp + 16 + lh * 8);
            accO[dt] = __builtin_amdgcn_wmma_f32_16x16x32_bf16(false, pa.v, false, bv.v, (short)0, accO[dt], false, false);
        }
    }
    #pragma unroll
    for (int r = 0; r < 8; ++r) {
        int qrow = qbase + lh * 8 + r;
        float inv = 1.0f / lrow[r];
        #pragma unroll
        for (int dt = 0; dt < 4; ++dt) {
            int d = dt * 16 + lm;
            attnOut[((size_t)(b * NT + qrow)) * C_DIM + h * DH + d] = f2bf(accO[dt][r] * inv);
        }
    }
}

// ---------------------------------------------------------------------------
// out = xres + gate(b,c) * y   (gated residual)
// ---------------------------------------------------------------------------
__global__ void mmd_addgate(
    float* __restrict__ out, const float* __restrict__ xres,
    const float* __restrict__ ada, int gateOff,
    const float* __restrict__ y, int yTokPerB, int yTokOff,
    int tokPerB, int total) {
    int idx = blockIdx.x * 256 + threadIdx.x;
    if (idx >= total) return;
    int cc = idx & (C_DIM - 1);
    int rown = idx >> 10;
    int b = rown / tokPerB;
    int n = rown % tokPerB;
    float g  = ada[(size_t)b * SIXC + gateOff + cc];
    float yv = y[((size_t)(b * yTokPerB + yTokOff + n)) * C_DIM + cc];
    out[idx] = xres[idx] + g * yv;
}

// ---------------------------------------------------------------------------
// SwiGLU: h[m][j] = silu(x12[m][j]) * x12[m][HID+j]   (padded K out)
// ---------------------------------------------------------------------------
__global__ void mmd_swiglu(const unsigned short* __restrict__ x12,
                           unsigned short* __restrict__ h, int M) {
    int idx = blockIdx.x * 256 + threadIdx.x;
    int total = M * HIDP;
    if (idx >= total) return;
    int j = idx % HIDP;
    int m = idx / HIDP;
    if (j >= HID) { h[(size_t)m * HIDP + j] = 0; return; }
    float x1 = bf2f(x12[(size_t)m * N2P + j]);
    float x2 = bf2f(x12[(size_t)m * N2P + HID + j]);
    float v = x1 / (1.0f + __expf(-x1)) * x2;
    h[(size_t)m * HIDP + j] = f2bf(v);
}

// ---------------------------------------------------------------------------
extern "C" void kernel_launch(void* const* d_in, const int* in_sizes, int n_in,
                              void* d_out, int out_size, void* d_ws, size_t ws_size,
                              hipStream_t stream) {
    const float* x_pixel = (const float*)d_in[0];
    const float* x_dino  = (const float*)d_in[1];
    const float* c       = (const float*)d_in[2];
    const float* w_ada_p = (const float*)d_in[3];
    const float* b_ada_p = (const float*)d_in[4];
    const float* w_ada_d = (const float*)d_in[5];
    const float* b_ada_d = (const float*)d_in[6];
    const float* g_n1p   = (const float*)d_in[7];
    const float* g_n1d   = (const float*)d_in[8];
    const float* g_n2p   = (const float*)d_in[9];
    const float* g_n2d   = (const float*)d_in[10];
    const float* w_qkv   = (const float*)d_in[11];
    const float* b_qkv   = (const float*)d_in[12];
    const float* g_qn    = (const float*)d_in[13];
    const float* g_kn    = (const float*)d_in[14];
    const float* w_proj  = (const float*)d_in[15];
    const float* b_proj  = (const float*)d_in[16];
    const float* w12_p   = (const float*)d_in[17];
    const float* b12_p   = (const float*)d_in[18];
    const float* w3_p    = (const float*)d_in[19];
    const float* b3_p    = (const float*)d_in[20];
    const float* w12_d   = (const float*)d_in[21];
    const float* b12_d   = (const float*)d_in[22];
    const float* w3_d    = (const float*)d_in[23];
    const float* b3_d    = (const float*)d_in[24];
    const unsigned char* bflag = (const unsigned char*)d_in[25];

    float* out = (float*)d_out;
    const int B = 4;
    const size_t pixTotal = (size_t)B * NP * C_DIM;

    char* ws = (char*)d_ws;
    size_t off = 0;
    auto alloc = [&](size_t bytes) { size_t r = off; off = (off + bytes + 255) & ~(size_t)255; return r; };

    float* ada_p   = (float*)(ws + alloc((size_t)B * SIXC * 4));
    float* ada_d   = (float*)(ws + alloc((size_t)B * SIXC * 4));
    unsigned short* wqkvT = (unsigned short*)(ws + alloc((size_t)3072 * 1024 * 2));
    unsigned short* wprojT= (unsigned short*)(ws + alloc((size_t)1024 * 1024 * 2));
    unsigned short* w12pT = (unsigned short*)(ws + alloc((size_t)N2P * 1024 * 2));
    unsigned short* w12dT = (unsigned short*)(ws + alloc((size_t)N2P * 1024 * 2));
    unsigned short* w3pT  = (unsigned short*)(ws + alloc((size_t)1024 * HIDP * 2));
    unsigned short* w3dT  = (unsigned short*)(ws + alloc((size_t)1024 * HIDP * 2));
    unsigned short* xc    = (unsigned short*)(ws + alloc((size_t)B * NT * C_DIM * 2));
    float* qkvF           = (float*)(ws + alloc((size_t)B * NT * 3 * C_DIM * 4));
    unsigned short* qbuf  = (unsigned short*)(ws + alloc((size_t)B * HEADS * NT * DH * 2));
    unsigned short* kbuf  = (unsigned short*)(ws + alloc((size_t)B * HEADS * NT * DH * 2));
    unsigned short* vbuf  = (unsigned short*)(ws + alloc((size_t)B * HEADS * DH * NT * 2));
    unsigned short* attnb = (unsigned short*)(ws + alloc((size_t)B * NT * C_DIM * 2));
    float* projF          = (float*)(ws + alloc((size_t)B * NT * C_DIM * 4));
    unsigned short* nmp   = (unsigned short*)(ws + alloc((size_t)B * NP * C_DIM * 2));
    unsigned short* nmd   = (unsigned short*)(ws + alloc((size_t)B * 256 * C_DIM * 2));
    unsigned short* x12p  = (unsigned short*)(ws + alloc((size_t)B * NP * N2P * 2));
    unsigned short* x12d  = (unsigned short*)(ws + alloc((size_t)B * 256 * N2P * 2));
    unsigned short* hp    = (unsigned short*)(ws + alloc((size_t)B * NP * HIDP * 2));
    unsigned short* hd    = (unsigned short*)(ws + alloc((size_t)B * 256 * HIDP * 2));
    float* mlp2p          = (float*)(ws + alloc((size_t)B * NP * C_DIM * 4));
    float* mlp2d          = (float*)(ws + alloc((size_t)B * 256 * C_DIM * 4));
    (void)ws_size; (void)in_sizes; (void)n_in; (void)out_size;

    auto cdiv = [](size_t a, size_t b) { return (unsigned)((a + b - 1) / b); };

    // 1. weight conversion (transpose + pad to bf16)
    mmd_convw<<<cdiv((size_t)3072 * 1024, 256), 256, 0, stream>>>(w_qkv,  wqkvT, 3072, 1024, 3072, 1024);
    mmd_convw<<<cdiv((size_t)1024 * 1024, 256), 256, 0, stream>>>(w_proj, wprojT, 1024, 1024, 1024, 1024);
    mmd_convw<<<cdiv((size_t)N2P * 1024, 256), 256, 0, stream>>>(w12_p, w12pT, N2, 1024, N2P, 1024);
    mmd_convw<<<cdiv((size_t)N2P * 1024, 256), 256, 0, stream>>>(w12_d, w12dT, N2, 1024, N2P, 1024);
    mmd_convw<<<cdiv((size_t)1024 * HIDP, 256), 256, 0, stream>>>(w3_p, w3pT, 1024, HID, 1024, HIDP);
    mmd_convw<<<cdiv((size_t)1024 * HIDP, 256), 256, 0, stream>>>(w3_d, w3dT, 1024, HID, 1024, HIDP);

    // 2. adaLN params
    mmd_ada<<<dim3(48, 4), 256, 0, stream>>>(c, w_ada_p, b_ada_p, w_ada_d, b_ada_d, ada_p, ada_d);

    // 3. norm1 + modulate -> xc bf16 [B,1280,C]
    mmd_normmod<<<B * NP, 256, 0, stream>>>(x_pixel, NP, g_n1p, ada_p, 0, 1024, xc, NT, 0);
    mmd_normmod<<<B * 256, 256, 0, stream>>>(x_dino, 256, g_n1d, ada_d, 0, 1024, xc, NT, NP);

    // 4. QKV GEMM [5120 x 3072 x 1024]
    mmd_gemm<<<dim3(24, 40), 256, 0, stream>>>(xc, wqkvT, b_qkv, qkvF, nullptr, 5120, 3072, 1024, 3072);

    // 5. qk-norm + RoPE + v-transpose
    mmd_qkv_prep<<<B * HEADS * NT, 32, 0, stream>>>(qkvF, g_qn, g_kn, qbuf, kbuf, vbuf);

    // 6. attention
    mmd_attn<<<dim3(10, HEADS, B), 256, 0, stream>>>(qbuf, kbuf, vbuf, bflag, attnb);

    // 7. output projection [5120 x 1024 x 1024]
    mmd_gemm<<<dim3(8, 40), 256, 0, stream>>>(attnb, wprojT, b_proj, projF, nullptr, 5120, 1024, 1024, 1024);

    // 8. gated residual -> d_out (x_mid)
    mmd_addgate<<<cdiv(pixTotal, 256), 256, 0, stream>>>(out, x_pixel, ada_p, 2048, projF, NT, 0, NP, (int)pixTotal);
    mmd_addgate<<<cdiv((size_t)B * 256 * C_DIM, 256), 256, 0, stream>>>(out + pixTotal, x_dino, ada_d, 2048, projF, NT, NP, 256, B * 256 * C_DIM);

    // 9. norm2 + modulate -> MLP inputs
    mmd_normmod<<<B * NP, 256, 0, stream>>>(out, NP, g_n2p, ada_p, 3072, 4096, nmp, NP, 0);
    mmd_normmod<<<B * 256, 256, 0, stream>>>(out + pixTotal, 256, g_n2d, ada_d, 3072, 4096, nmd, 256, 0);

    // 10. MLP GEMM1 [M x 5460 x 1024] -> bf16 (padded ldc)
    mmd_gemm<<<dim3(43, 32), 256, 0, stream>>>(nmp, w12pT, b12_p, nullptr, x12p, 4096, N2, 1024, N2P);
    mmd_gemm<<<dim3(43, 8), 256, 0, stream>>>(nmd, w12dT, b12_d, nullptr, x12d, 1024, N2, 1024, N2P);

    // 11. SwiGLU -> hidden bf16 (K padded with zeros)
    mmd_swiglu<<<cdiv((size_t)4096 * HIDP, 256), 256, 0, stream>>>(x12p, hp, 4096);
    mmd_swiglu<<<cdiv((size_t)1024 * HIDP, 256), 256, 0, stream>>>(x12d, hd, 1024);

    // 12. MLP GEMM2 [M x 1024 x 2752]
    mmd_gemm<<<dim3(8, 32), 256, 0, stream>>>(hp, w3pT, b3_p, mlp2p, nullptr, 4096, 1024, HIDP, 1024);
    mmd_gemm<<<dim3(8, 8), 256, 0, stream>>>(hd, w3dT, b3_d, mlp2d, nullptr, 1024, 1024, HIDP, 1024);

    // 13. final gated residual (in-place on d_out)
    mmd_addgate<<<cdiv(pixTotal, 256), 256, 0, stream>>>(out, out, ada_p, 5120, mlp2p, NP, 0, NP, (int)pixTotal);
    mmd_addgate<<<cdiv((size_t)B * 256 * C_DIM, 256), 256, 0, stream>>>(out + pixTotal, out + pixTotal, ada_d, 5120, mlp2d, 256, 0, 256, B * 256 * C_DIM);
}
